// PhyloformerSeq_23081154248856
// MI455X (gfx1250) — matrix operand order repaired and checked by reading, not compiled, not run on hardware
//
#include <hip/hip_runtime.h>
#include <hip/hip_fp16.h>
#include <math.h>

typedef _Float16 f16;
typedef __attribute__((ext_vector_type(4)))  _Float16 v4h;
typedef __attribute__((ext_vector_type(8)))  _Float16 v8h;
typedef __attribute__((ext_vector_type(16))) _Float16 v16h;
typedef __attribute__((ext_vector_type(8)))  float    v8f;

#define B_   2
#define D_   128
#define SL_  1024
#define N_   64
#define H_   4
#define LL_  4
#define M_TOK (B_*SL_*N_)   // 131072 tokens

// ---------------------------------------------------------------------------
// x [B, D, SL, N] -> y [B, SL, N, D] (fp32), LDS 32x32 tile transpose
// ---------------------------------------------------------------------------
__global__ void transpose_in(const float* __restrict__ x, float* __restrict__ y) {
  __shared__ float t[32][33];
  int bs   = blockIdx.x;            // b*SL + sl
  int tile = blockIdx.y;            // 0..7 : 4 d-tiles x 2 n-tiles
  int td = (tile >> 1) * 32;
  int tn = (tile & 1) * 32;
  int b = bs >> 10, sl = bs & 1023;
  int tx = threadIdx.x, ty = threadIdx.y;
#pragma unroll
  for (int j = 0; j < 4; ++j) {
    int d = td + ty + j * 8;
    t[ty + j * 8][tx] = x[(((size_t)b * D_ + d) * SL_ + sl) * N_ + tn + tx];
  }
  __syncthreads();
#pragma unroll
  for (int j = 0; j < 4; ++j) {
    int n = tn + ty + j * 8;
    y[((size_t)bs * N_ + n) * D_ + td + tx] = t[tx][ty + j * 8];
  }
}

// ---------------------------------------------------------------------------
// LayerNorm(row of 128) * g + b -> f16.  One wave32 per row, 4 elems/lane.
// ---------------------------------------------------------------------------
__global__ void ln_to_f16(const float* __restrict__ y, const float* __restrict__ g,
                          const float* __restrict__ bta, f16* __restrict__ out) {
  int lane = threadIdx.x & 31;
  int row  = blockIdx.x * 8 + (threadIdx.x >> 5);
  const float4 v = *(const float4*)(y + (size_t)row * D_ + lane * 4);
  float s  = v.x + v.y + v.z + v.w;
  float ss = v.x * v.x + v.y * v.y + v.z * v.z + v.w * v.w;
#pragma unroll
  for (int m = 16; m >= 1; m >>= 1) {
    s  += __shfl_xor(s, m, 32);
    ss += __shfl_xor(ss, m, 32);
  }
  float mean = s * (1.0f / D_);
  float var  = ss * (1.0f / D_) - mean * mean;
  float rstd = rsqrtf(var + 1e-5f);
  const float4 gg = *(const float4*)(g + lane * 4);
  const float4 bb = *(const float4*)(bta + lane * 4);
  v4h o;
  o[0] = (f16)((v.x - mean) * rstd * gg.x + bb.x);
  o[1] = (f16)((v.y - mean) * rstd * gg.y + bb.y);
  o[2] = (f16)((v.z - mean) * rstd * gg.z + bb.z);
  o[3] = (f16)((v.w - mean) * rstd * gg.w + bb.w);
  *(v4h*)(out + (size_t)row * D_ + lane * 4) = o;
}

// ---------------------------------------------------------------------------
// WMMA GEMM: C[M x Nout] = A[M x K](f16) * W[Nout x K](f16)^T + bias
// MODE 0: store f16    MODE 1: exact GELU then store f16    MODE 2: resid += C
// Block: 256 thr = 8 waves; wave tile 16x64 (A reused over 4 wmma); block 128x64.
// Fragment layouts per CDNA5 ISA (16x32 f16 A, 32x16 B, 16x16 f32 D).
// ---------------------------------------------------------------------------
template <int K, int MODE>
__global__ void gemm_wmma(const f16* __restrict__ A, const f16* __restrict__ W,
                          const float* __restrict__ bias, f16* __restrict__ out16,
                          float* __restrict__ resid, int Nout) {
  int lane = threadIdx.x & 31;
  int w    = threadIdx.x >> 5;
  int hi   = lane >> 4;
  int ln15 = lane & 15;
  int rowbase = blockIdx.y * 128 + w * 16;
  int colbase = blockIdx.x * 64;
  int r = rowbase + ln15;

  v8f cacc[4];
#pragma unroll
  for (int j = 0; j < 4; ++j) cacc[j] = (v8f){0.f,0.f,0.f,0.f,0.f,0.f,0.f,0.f};

  const f16* arow = A + (size_t)r * K;
#pragma unroll 4
  for (int kk = 0; kk < K; kk += 32) {
    // A 16x32: lanes 0-15 hold K kk..kk+15; lanes 16-31 hold kk+16..kk+31 (contiguous, matches B)
    v8h alo = *(const v8h*)(arow + kk + hi * 16);
    v8h ahi = *(const v8h*)(arow + kk + hi * 16 + 8);
    v16h a;
#pragma unroll
    for (int i = 0; i < 8; ++i) { a[i] = alo[i]; a[i + 8] = ahi[i]; }
#pragma unroll
    for (int j = 0; j < 4; ++j) {
      int n = colbase + j * 16 + ln15;
      // B 32x16 (= W^T): lane&15 = column n; lane group selects K half; contiguous 16 halves
      v16h bf = *(const v16h*)(W + (size_t)n * K + kk + hi * 16);
      cacc[j] = __builtin_amdgcn_wmma_f32_16x16x32_f16(
          false, a, false, bf, (short)0, cacc[j], false, false);
    }
  }

#pragma unroll
  for (int j = 0; j < 4; ++j) {
    int n = colbase + j * 16 + ln15;
    float bsv = bias[n];
#pragma unroll
    for (int i = 0; i < 8; ++i) {
      int m = rowbase + hi * 8 + i;           // D layout: VGPR i -> M = i (+8 for hi lanes)
      float val = cacc[j][i] + bsv;
      if (MODE == 1) val = 0.5f * val * (1.0f + erff(val * 0.70710678118f));
      if (MODE == 2) {
        size_t off = (size_t)m * Nout + n;
        resid[off] += val;
      } else {
        out16[(size_t)m * Nout + n] = (f16)val;
      }
    }
  }
}

// ---------------------------------------------------------------------------
// Rank-1 scaled linear attention combine, row groups (C = N = 64 tokens, contiguous)
// Per group: q/k = elu(h @ w^T + b)+1; q /= mean_C; k /= sum_C; ktv = sum_C k*v;
// out[c, h*32+e] = q[c,h] * ktv[h,e]
// ---------------------------------------------------------------------------
__global__ void row_combine(const f16* __restrict__ h16, const f16* __restrict__ v16,
                            const float* __restrict__ qw, const float* __restrict__ qb,
                            const float* __restrict__ kw, const float* __restrict__ kb,
                            f16* __restrict__ vout) {
  __shared__ float qs_[N_][H_];
  __shared__ float ks_[N_][H_];
  __shared__ float sums[2][H_];
  int t = threadIdx.x;                       // 128 threads
  size_t base = (size_t)blockIdx.x * N_;     // tokens base..base+63
#pragma unroll
  for (int it = 0; it < 4; ++it) {           // 512 dots total
    int idx = t + it * 128;
    int c = idx >> 3, rr = idx & 7, h = rr >> 1, isK = rr & 1;
    const float* wv = (isK ? kw : qw) + h * D_;
    const f16*   hr = h16 + (base + c) * D_;
    float acc = (isK ? kb : qb)[h];
    for (int d = 0; d < D_; ++d) acc += (float)hr[d] * wv[d];
    acc = acc > 0.f ? acc + 1.f : __expf(acc);   // elu(x)+1
    (isK ? ks_ : qs_)[c][h] = acc;
  }
  __syncthreads();
  if (t < 8) {
    int h = t & 3, isK = t >> 2;
    float s = 0.f;
    for (int c = 0; c < N_; ++c) s += (isK ? ks_ : qs_)[c][h];
    sums[isK][h] = s;
  }
  __syncthreads();
  int h = t >> 5;                            // t = h*32 + e
  float kt = 0.f;
  for (int c = 0; c < N_; ++c) kt += ks_[c][h] * (float)v16[(base + c) * D_ + t];
  kt /= sums[1][h];
  float invqm = (float)N_ / sums[0][h];
  for (int c = 0; c < N_; ++c)
    vout[(base + c) * D_ + t] = (f16)(qs_[c][h] * invqm * kt);
}

// Column groups: (b, n), C = SL = 1024 tokens strided by N*D
__global__ void col_combine(const f16* __restrict__ h16, const f16* __restrict__ v16,
                            const float* __restrict__ qw, const float* __restrict__ qb,
                            const float* __restrict__ kw, const float* __restrict__ kb,
                            f16* __restrict__ vout) {
  __shared__ float qs_[SL_][H_];             // 16 KB
  __shared__ float ks_[SL_][H_];             // 16 KB
  __shared__ float part[8][16];
  __shared__ float sums[2][H_];
  int t = threadIdx.x;                       // 128 threads
  int b = blockIdx.x >> 6, n = blockIdx.x & 63;
  size_t base = (size_t)b * SL_ * N_ + n;    // token(c) = base + c*N_
  for (int it = 0; it < 64; ++it) {          // 8192 dots total
    int idx = t + it * 128;
    int c = idx >> 3, rr = idx & 7, h = rr >> 1, isK = rr & 1;
    const float* wv = (isK ? kw : qw) + h * D_;
    const f16*   hr = h16 + (base + (size_t)c * N_) * D_;
    float acc = (isK ? kb : qb)[h];
    for (int d = 0; d < D_; ++d) acc += (float)hr[d] * wv[d];
    acc = acc > 0.f ? acc + 1.f : __expf(acc);
    (isK ? ks_ : qs_)[c][h] = acc;
  }
  __syncthreads();
  {
    int g = t >> 4, s = t & 15;              // 8 groups (isK,h) x 16 partials
    int isK = g >> 2, h = g & 3;
    float p = 0.f;
    for (int c = s; c < SL_; c += 16) p += (isK ? ks_ : qs_)[c][h];
    part[g][s] = p;
  }
  __syncthreads();
  if (t < 8) {
    float sm = 0.f;
    for (int i = 0; i < 16; ++i) sm += part[t][i];
    sums[t >> 2][t & 3] = sm;
  }
  __syncthreads();
  int h = t >> 5;
  float kt = 0.f;
  for (int c = 0; c < SL_; ++c)
    kt += ks_[c][h] * (float)v16[(base + (size_t)c * N_) * D_ + t];
  kt /= sums[1][h];
  float invqm = (float)SL_ / sums[0][h];
  for (int c = 0; c < SL_; ++c)
    vout[(base + (size_t)c * N_) * D_ + t] = (f16)(qs_[c][h] * invqm * kt);
}

// ---------------------------------------------------------------------------
// xb[b,n,d] = mean over sl=1..SL-1 of y
// ---------------------------------------------------------------------------
__global__ void mean_sites(const float* __restrict__ y, float* __restrict__ xb) {
  int t = threadIdx.x;                       // d
  int b = blockIdx.x >> 6, n = blockIdx.x & 63;
  float acc = 0.f;
  for (int sl = 1; sl < SL_; ++sl)
    acc += y[(((size_t)b * SL_ + sl) * N_ + n) * D_ + t];
  xb[((size_t)b * N_ + n) * D_ + t] = acc * (1.0f / (SL_ - 1));
}

// One wave32 per (b, pair) lower-triangular distance, np.tril_indices order
__global__ void pairdist(const float* __restrict__ xb, float* __restrict__ out) {
  const int P = N_ * (N_ - 1) / 2;           // 2016
  int lane = threadIdx.x & 31;
  int gid  = blockIdx.x * 8 + (threadIdx.x >> 5);
  if (gid >= B_ * P) return;
  int b = gid / P, p = gid % P;
  int i = (int)((1.0f + sqrtf(1.0f + 8.0f * (float)p)) * 0.5f);
  while (i * (i - 1) / 2 > p) --i;
  while ((i + 1) * i / 2 <= p) ++i;
  int j = p - i * (i - 1) / 2;
  const float* xi = xb + ((size_t)b * N_ + i) * D_;
  const float* xj = xb + ((size_t)b * N_ + j) * D_;
  float acc = 0.f;
#pragma unroll
  for (int q = 0; q < 4; ++q) {
    float d = xi[lane * 4 + q] - xj[lane * 4 + q];
    acc += d * d;
  }
#pragma unroll
  for (int m = 16; m >= 1; m >>= 1) acc += __shfl_xor(acc, m, 32);
  if (lane == 0) out[(size_t)b * P + p] = sqrtf(acc + 1e-8f);
}

__global__ void cvt_f16(const float* __restrict__ src, f16* __restrict__ dst, int n) {
  int i = blockIdx.x * blockDim.x + threadIdx.x;
  if (i < n) dst[i] = (f16)src[i];
}

// ---------------------------------------------------------------------------
extern "C" void kernel_launch(void* const* d_in, const int* in_sizes, int n_in,
                              void* d_out, int out_size, void* d_ws, size_t ws_size,
                              hipStream_t stream) {
  const float* x    = (const float*)d_in[0];
  const float* rq_w = (const float*)d_in[1];  const float* rq_b = (const float*)d_in[2];
  const float* rk_w = (const float*)d_in[3];  const float* rk_b = (const float*)d_in[4];
  const float* rv_w = (const float*)d_in[5];  const float* rv_b = (const float*)d_in[6];
  const float* ro_w = (const float*)d_in[7];  const float* ro_b = (const float*)d_in[8];
  const float* cq_w = (const float*)d_in[9];  const float* cq_b = (const float*)d_in[10];
  const float* ck_w = (const float*)d_in[11]; const float* ck_b = (const float*)d_in[12];
  const float* cv_w = (const float*)d_in[13]; const float* cv_b = (const float*)d_in[14];
  const float* co_w = (const float*)d_in[15]; const float* co_b = (const float*)d_in[16];
  const float* lrg  = (const float*)d_in[17]; const float* lrb  = (const float*)d_in[18];
  const float* lcg  = (const float*)d_in[19]; const float* lcb  = (const float*)d_in[20];
  const float* lfg  = (const float*)d_in[21]; const float* lfb  = (const float*)d_in[22];
  const float* f1_w = (const float*)d_in[23]; const float* f1_b = (const float*)d_in[24];
  const float* f2_w = (const float*)d_in[25]; const float* f2_b = (const float*)d_in[26];

  // workspace layout (all 256B aligned)
  char* ws = (char*)d_ws;
  float* y    = (float*)(ws);                                       // 64 MB fp32 activations
  f16*   h16  = (f16*)(ws + 67108864ull);                           // 32 MB LN output
  f16*   v16  = (f16*)(ws + 67108864ull + 33554432ull);             // 32 MB v projection
  f16*   vo16 = (f16*)(ws + 67108864ull + 2ull * 33554432ull);      // 32 MB attn output
  f16*   hf16 = (f16*)(ws + 67108864ull + 3ull * 33554432ull);      // 128 MB FFN hidden
  char*  p5   = ws + 67108864ull + 3ull * 33554432ull + 134217728ull;
  float* xb   = (float*)(p5);                                       // 64 KB site means
  f16*   w_rv = (f16*)(p5 + 65536);
  f16*   w_ro = (f16*)(p5 + 65536 + 1 * 131072);
  f16*   w_cv = (f16*)(p5 + 65536 + 2 * 131072);
  f16*   w_co = (f16*)(p5 + 65536 + 3 * 131072);
  f16*   w_f1 = (f16*)(p5 + 65536 + 4 * 131072);
  f16*   w_f2 = (f16*)(p5 + 65536 + 4 * 131072 + 524288);

  const int WD = D_ * D_, WF = 4 * D_ * D_;
  cvt_f16<<<(LL_ * WD + 255) / 256, 256, 0, stream>>>(rv_w, w_rv, LL_ * WD);
  cvt_f16<<<(LL_ * WD + 255) / 256, 256, 0, stream>>>(ro_w, w_ro, LL_ * WD);
  cvt_f16<<<(LL_ * WD + 255) / 256, 256, 0, stream>>>(cv_w, w_cv, LL_ * WD);
  cvt_f16<<<(LL_ * WD + 255) / 256, 256, 0, stream>>>(co_w, w_co, LL_ * WD);
  cvt_f16<<<(LL_ * WF + 255) / 256, 256, 0, stream>>>(f1_w, w_f1, LL_ * WF);
  cvt_f16<<<(LL_ * WF + 255) / 256, 256, 0, stream>>>(f2_w, w_f2, LL_ * WF);

  transpose_in<<<dim3(B_ * SL_, 8), dim3(32, 8), 0, stream>>>(x, y);

  const dim3 gGemmD(2, M_TOK / 128);   // Nout=128
  const dim3 gGemmF(8, M_TOK / 128);   // Nout=512
  for (int l = 0; l < LL_; ++l) {
    // ---- row attention ----
    ln_to_f16<<<M_TOK / 8, 256, 0, stream>>>(y, lrg + l * D_, lrb + l * D_, h16);
    gemm_wmma<128, 0><<<gGemmD, 256, 0, stream>>>(h16, w_rv + l * WD, rv_b + l * D_, v16, nullptr, D_);
    row_combine<<<B_ * SL_, 128, 0, stream>>>(h16, v16, rq_w + l * H_ * D_, rq_b + l * H_,
                                              rk_w + l * H_ * D_, rk_b + l * H_, vo16);
    gemm_wmma<128, 2><<<gGemmD, 256, 0, stream>>>(vo16, w_ro + l * WD, ro_b + l * D_, nullptr, y, D_);
    // ---- column attention ----
    ln_to_f16<<<M_TOK / 8, 256, 0, stream>>>(y, lcg + l * D_, lcb + l * D_, h16);
    gemm_wmma<128, 0><<<gGemmD, 256, 0, stream>>>(h16, w_cv + l * WD, cv_b + l * D_, v16, nullptr, D_);
    col_combine<<<B_ * N_, 128, 0, stream>>>(h16, v16, cq_w + l * H_ * D_, cq_b + l * H_,
                                             ck_w + l * H_ * D_, ck_b + l * H_, vo16);
    gemm_wmma<128, 2><<<gGemmD, 256, 0, stream>>>(vo16, w_co + l * WD, co_b + l * D_, nullptr, y, D_);
    // ---- FFN ----
    ln_to_f16<<<M_TOK / 8, 256, 0, stream>>>(y, lfg + l * D_, lfb + l * D_, h16);
    gemm_wmma<128, 1><<<gGemmF, 256, 0, stream>>>(h16, w_f1 + l * WF, f1_b + l * 4 * D_, hf16, nullptr, 4 * D_);
    gemm_wmma<512, 2><<<gGemmD, 256, 0, stream>>>(hf16, w_f2 + l * WF, f2_b + l * D_, nullptr, y, D_);
  }

  mean_sites<<<B_ * N_, D_, 0, stream>>>(y, xb);
  pairdist<<<(B_ * 2016 + 7) / 8, 256, 0, stream>>>(xb, (float*)d_out);
}
